// Head_24713241821863
// MI455X (gfx1250) — matrix operand order repaired and checked by reading, not compiled, not run on hardware
//
#include <hip/hip_runtime.h>

typedef __attribute__((ext_vector_type(16))) _Float16 v16h;
typedef __attribute__((ext_vector_type(8)))  _Float16 v8h;
typedef __attribute__((ext_vector_type(8)))  float    v8f;

union V16H { v16h v; v8h h[2]; };

#define NUM_EMBED 1024
#define HEAD_SIZE 64
#define BATCH 4
#define SEQ 4096
#define MROWS (BATCH * SEQ)          // 16384
#define MTILES (MROWS / 16)          // 1024 row tiles

// ---------------------------------------------------------------------------
// Kernel 0: pack Wq/Wk/Wv (f32 [1024][64]) into f16 WMMA B-operand layout.
// Packed index: (((w*4 + nt)*32 + kc)*32 + lane)*16 + i
//   lane<16 : col n = nt*16+lane,    K = kc*32 + {0..7,16..23}[i]
//   lane>=16: col n = nt*16+lane-16, K = kc*32 + {8..15,24..31}[i]
// 1/sqrt(HEAD_SIZE) folded into Wq.
// ---------------------------------------------------------------------------
__global__ void pack_w(const float* __restrict__ Wq, const float* __restrict__ Wk,
                       const float* __restrict__ Wv, _Float16* __restrict__ Bpack) {
    int idx = blockIdx.x * blockDim.x + threadIdx.x;
    if (idx >= 3 * 65536) return;
    int w    = idx >> 16;
    int i    = idx & 15;
    int lane = (idx >> 4) & 31;
    int kc   = (idx >> 9) & 31;
    int nt   = (idx >> 14) & 3;
    const float* W = (w == 0) ? Wq : ((w == 1) ? Wk : Wv);
    int k = kc * 32 + ((lane >= 16) ? 8 : 0) + ((i < 8) ? i : i + 8);
    int n = nt * 16 + (lane & 15);
    float val = W[k * HEAD_SIZE + n];
    if (w == 0) val *= 0.125f;   // HEAD_SIZE^-0.5
    Bpack[idx] = (_Float16)val;
}

__device__ inline v8h cvt8(float4 a, float4 b) {
    v8h r;
    r[0] = (_Float16)a.x; r[1] = (_Float16)a.y; r[2] = (_Float16)a.z; r[3] = (_Float16)a.w;
    r[4] = (_Float16)b.x; r[5] = (_Float16)b.y; r[6] = (_Float16)b.z; r[7] = (_Float16)b.w;
    return r;
}

__device__ inline v8f wmma_f16(v16h a, v16h b, v8f c) {
    return __builtin_amdgcn_wmma_f32_16x16x32_f16(false, a, false, b, (short)0, c, false, false);
}

// ---------------------------------------------------------------------------
// Kernel 1: fused q/k/v projection. One wave computes the 16x64 tiles of ALL
// THREE projections for its row tile: x is streamed once (64 MB total instead
// of 192 MB), 12 WMMAs per 32-deep k-chunk. B tiles batched into registers
// per weight before the WMMAs so loads overlap the matrix pipe.
// q,k stored f16 row-major [16384][64]; v stored transposed f16 [b][64][4096].
// ---------------------------------------------------------------------------
__global__ __launch_bounds__(128) void proj_kernel(
        const float* __restrict__ x, const _Float16* __restrict__ Bpack,
        _Float16* __restrict__ qf, _Float16* __restrict__ kf, _Float16* __restrict__ vT) {
    int wave = (blockIdx.x * blockDim.x + threadIdx.x) >> 5;   // 0..1023
    int lane = threadIdx.x & 31;
    int t0 = wave * 16;
    int hi  = lane >> 4;
    int l16 = lane & 15;

    v8f acc[3][4] = {};
    const float* xrow = x + (size_t)(t0 + l16) * NUM_EMBED;
    int abase = hi ? 8 : 0;   // A layout: lane<16 K={0..7,16..23}, lane>=16 K={8..15,24..31}

    for (int kc = 0; kc < 32; ++kc) {
        int kb = kc * 32 + abase;
        float4 xa0 = *(const float4*)(xrow + kb);
        float4 xa1 = *(const float4*)(xrow + kb + 4);
        float4 xa2 = *(const float4*)(xrow + kb + 16);
        float4 xa3 = *(const float4*)(xrow + kb + 20);
        V16H a;
        a.h[0] = cvt8(xa0, xa1);
        a.h[1] = cvt8(xa2, xa3);

        #pragma unroll
        for (int w = 0; w < 3; ++w) {
            const _Float16* bp = Bpack + (((size_t)(w * 4) * 32 + kc) * 32 + lane) * 16;
            V16H b[4];
            #pragma unroll
            for (int nt = 0; nt < 4; ++nt) {
                const _Float16* bb = bp + (size_t)nt * (32 * 32 * 16);
                b[nt].h[0] = *(const v8h*)(bb);
                b[nt].h[1] = *(const v8h*)(bb + 8);
            }
            #pragma unroll
            for (int nt = 0; nt < 4; ++nt)
                acc[w][nt] = wmma_f16(a.v, b[nt].v, acc[w][nt]);
        }
    }

    int mbase = t0 + (hi ? 8 : 0);
    // q, k: row-major f16
    #pragma unroll
    for (int nt = 0; nt < 4; ++nt)
        #pragma unroll
        for (int r = 0; r < 8; ++r) {
            qf[(size_t)(mbase + r) * HEAD_SIZE + nt * 16 + l16] = (_Float16)acc[0][nt][r];
            kf[(size_t)(mbase + r) * HEAD_SIZE + nt * 16 + l16] = (_Float16)acc[1][nt][r];
        }
    // v: transposed per batch [b][h][t]
    #pragma unroll
    for (int nt = 0; nt < 4; ++nt)
        #pragma unroll
        for (int r = 0; r < 8; ++r) {
            int t  = mbase + r;
            int bb = t >> 12;
            int tt = t & 4095;
            int h  = nt * 16 + l16;
            vT[(((size_t)(bb * HEAD_SIZE + h)) << 12) + tt] = (_Float16)acc[2][nt][r];
        }
}

// ---------------------------------------------------------------------------
// Kernel 2: flash attention. One wave per 16-row query tile; key step = 64
// (4 score subtiles -> 8 score WMMAs + 8 PV WMMAs per iteration).
// ---------------------------------------------------------------------------
__global__ __launch_bounds__(128) void attn_kernel(
        const _Float16* __restrict__ qf, const _Float16* __restrict__ kf,
        const _Float16* __restrict__ vT, float* __restrict__ out) {
    __shared__ __align__(128) _Float16 Pstage[4][16 * 64];   // 2KB per wave

    int wslot = threadIdx.x >> 5;
    int lane  = threadIdx.x & 31;
    int qt = blockIdx.x * 4 + wslot;      // 0..1023
    int b  = qt >> 8;
    int t0 = (qt & 255) * 16;
    size_t rowbase = ((size_t)b) << 12;   // b*4096
    int hi  = lane >> 4;
    int l16 = lane & 15;
    _Float16* pst = Pstage[wslot];

    // q A-operand, resident for the whole key loop (h chunks 0-31 / 32-63)
    const _Float16* qrow = qf + (rowbase + t0 + l16) * HEAD_SIZE;
    int ha = hi ? 8 : 0;
    V16H qa0, qa1;
    qa0.h[0] = *(const v8h*)(qrow + ha);
    qa0.h[1] = *(const v8h*)(qrow + ha + 16);
    qa1.h[0] = *(const v8h*)(qrow + 32 + ha);
    qa1.h[1] = *(const v8h*)(qrow + 32 + ha + 16);

    float mrow[8], lrow[8];
    #pragma unroll
    for (int r = 0; r < 8; ++r) { mrow[r] = -1e30f; lrow[r] = 0.f; }
    v8f O[4] = {};

    for (int s0 = 0; s0 <= t0 + 15; s0 += 64) {
        // ---- scores: four 16-col subtiles, K=64 contraction via 2 WMMAs each
        v8f S[4];
        #pragma unroll
        for (int j = 0; j < 4; ++j) {
            const _Float16* krow = kf + (rowbase + s0 + 16 * j + l16) * HEAD_SIZE;
            V16H b0, b1;
            b0.h[0] = *(const v8h*)(krow + hi * 8);
            b0.h[1] = *(const v8h*)(krow + hi * 8 + 16);
            b1.h[0] = *(const v8h*)(krow + 32 + hi * 8);
            b1.h[1] = *(const v8h*)(krow + 32 + hi * 8 + 16);
            v8f z = {};
            z = wmma_f16(qa0.v, b0.v, z);
            z = wmma_f16(qa1.v, b1.v, z);
            S[j] = z;
        }

        bool need_mask = (s0 + 63 > t0);

        // ---- online softmax: combine 4 elems/lane, then 16-lane reductions
        float scale[8];
        #pragma unroll
        for (int r = 0; r < 8; ++r) {
            float v0 = S[0][r], v1 = S[1][r], v2 = S[2][r], v3 = S[3][r];
            if (need_mask) {
                int t = t0 + hi * 8 + r;
                if (s0 + l16 > t)      v0 = -1e30f;
                if (s0 + 16 + l16 > t) v1 = -1e30f;
                if (s0 + 32 + l16 > t) v2 = -1e30f;
                if (s0 + 48 + l16 > t) v3 = -1e30f;
                S[0][r] = v0; S[1][r] = v1; S[2][r] = v2; S[3][r] = v3;
            }
            float mx = fmaxf(fmaxf(v0, v1), fmaxf(v2, v3));
            #pragma unroll
            for (int off = 1; off < 16; off <<= 1)
                mx = fmaxf(mx, __shfl_xor(mx, off, 32));
            float nm = fmaxf(mrow[r], mx);
            scale[r] = __expf(mrow[r] - nm);
            mrow[r]  = nm;
        }
        #pragma unroll
        for (int r = 0; r < 8; ++r) {
            int m = hi * 8 + r;                        // C-layout row of this lane
            float rs = 0.f;
            #pragma unroll
            for (int j = 0; j < 4; ++j) {
                float p = __expf(S[j][r] - mrow[r]);
                rs += p;
                pst[m * 64 + 16 * j + l16] = (_Float16)p;
            }
            #pragma unroll
            for (int off = 1; off < 16; off <<= 1)
                rs += __shfl_xor(rs, off, 32);
            lrow[r] = lrow[r] * scale[r] + rs;
        }
        // same-wave LDS ordering: force stores complete & block compiler reorder
        asm volatile("s_wait_dscnt 0" ::: "memory");

        // ---- re-read P in A-operand layout (two K=32 chunks: s 0-31, 32-63)
        V16H pa0, pa1;
        {
            int sb = hi ? 8 : 0;
            const _Float16* pr = pst + l16 * 64;
            pa0.h[0] = *(const v8h*)(pr + sb);
            pa0.h[1] = *(const v8h*)(pr + sb + 16);
            pa1.h[0] = *(const v8h*)(pr + 32 + sb);
            pa1.h[1] = *(const v8h*)(pr + 32 + sb + 16);
        }
        asm volatile("" ::: "memory");

        // ---- P·V: rescale folded into first WMMA's C operand, chain K chunks
        #pragma unroll
        for (int nt = 0; nt < 4; ++nt) {
            const _Float16* vrow = vT + (((size_t)(b * HEAD_SIZE + nt * 16 + l16)) << 12)
                                      + s0 + hi * 8;
            V16H vb0, vb1;
            vb0.h[0] = *(const v8h*)(vrow);
            vb0.h[1] = *(const v8h*)(vrow + 16);
            vb1.h[0] = *(const v8h*)(vrow + 32);
            vb1.h[1] = *(const v8h*)(vrow + 48);
            v8f cin;
            #pragma unroll
            for (int r = 0; r < 8; ++r) cin[r] = O[nt][r] * scale[r];
            v8f tacc = wmma_f16(pa0.v, vb0.v, cin);
            O[nt] = wmma_f16(pa1.v, vb1.v, tacc);
        }
    }

    // ---- epilogue: out[b][t][h] = O / l
    #pragma unroll
    for (int nt = 0; nt < 4; ++nt)
        #pragma unroll
        for (int r = 0; r < 8; ++r) {
            int t = t0 + hi * 8 + r;
            out[(rowbase + t) * HEAD_SIZE + nt * 16 + l16] = O[nt][r] / lrow[r];
        }
}

// ---------------------------------------------------------------------------
extern "C" void kernel_launch(void* const* d_in, const int* in_sizes, int n_in,
                              void* d_out, int out_size, void* d_ws, size_t ws_size,
                              hipStream_t stream) {
    const float* x  = (const float*)d_in[0];
    const float* Wq = (const float*)d_in[1];
    const float* Wk = (const float*)d_in[2];
    const float* Wv = (const float*)d_in[3];

    char* ws = (char*)d_ws;
    _Float16* qf    = (_Float16*)(ws);                       // 2 MB
    _Float16* kf    = (_Float16*)(ws + (2u << 20));          // 2 MB
    _Float16* vT    = (_Float16*)(ws + (4u << 20));          // 2 MB
    _Float16* Bpack = (_Float16*)(ws + (6u << 20));          // 384 KB

    pack_w<<<768, 256, 0, stream>>>(Wq, Wk, Wv, Bpack);
    proj_kernel<<<256, 128, 0, stream>>>(x, Bpack, qf, kf, vT);
    attn_kernel<<<256, 128, 0, stream>>>(qf, kf, vT, (float*)d_out);
}
